// Decoder_48000554500451
// MI455X (gfx1250) — matrix operand order repaired and checked by reading, not compiled, not run on hardware
//
#include <hip/hip_runtime.h>
#include <hip/hip_bf16.h>
#include <cstdint>

#define T_STEPS 128
#define BB 512
#define SS 256
#define EE 128
#define HH 128
#define NEG_INF_F (-1e30f)

typedef __attribute__((ext_vector_type(16))) __bf16 v16bf;
typedef __attribute__((ext_vector_type(8)))  float  v8f;
typedef __attribute__((ext_vector_type(4)))  int    v4i_;

// ---------- CDNA5 async global->LDS support (guarded; falls back to ds_store) ----------
#if defined(__has_builtin)
#  if __has_builtin(__builtin_amdgcn_global_load_async_to_lds_b128)
#    define HAS_ASYNC_LDS 1
#  endif
#endif
#ifndef HAS_ASYNC_LDS
#  define HAS_ASYNC_LDS 0
#endif

#if HAS_ASYNC_LDS
__device__ __forceinline__ void async_copy_b128(const void* gsrc, void* ldst) {
  __builtin_amdgcn_global_load_async_to_lds_b128(
      (__attribute__((address_space(1))) v4i_*)(uintptr_t)gsrc,
      (__attribute__((address_space(3))) v4i_*)ldst,
      0, 0);
}
__device__ __forceinline__ void wait_asynccnt0() {
#  if __has_builtin(__builtin_amdgcn_s_wait_asynccnt)
  __builtin_amdgcn_s_wait_asynccnt(0);
#  else
  asm volatile("s_wait_asynccnt 0" ::: "memory");
#  endif
}
#endif

// ---------- helpers ----------

__device__ __forceinline__ unsigned short f2bf_bits(float f) {
  union { float f; unsigned u; } v; v.f = f;
  unsigned r = v.u + 0x7FFFu + ((v.u >> 16) & 1u);   // round-to-nearest-even
  return (unsigned short)(r >> 16);
}
__device__ __forceinline__ void store_bf(__bf16* dst, float f) {
  *(unsigned short*)dst = f2bf_bits(f);
}
__device__ __forceinline__ float sigmoidf(float x) {
  return 1.0f / (1.0f + __expf(-x));
}
__device__ __forceinline__ unsigned hash3(unsigned a, unsigned b, unsigned c) {
  unsigned h = a * 0x9E3779B1u ^ b * 0x85EBCA77u ^ c * 0xC2B2AE3Du;
  h ^= h >> 16; h *= 0x7FEB352Du; h ^= h >> 15; h *= 0x846CA68Bu; h ^= h >> 16;
  return h;
}

// ---------- WMMA bf16 GEMM: C[m][n] = sum_k A[m][k]*W[n][k] + bias[n] ----------
// A fragment, 16x32 bf16 (ISA 7.12.2): lane<16 -> M=lane, K = kk+{0..7, 16..23};
// lane>=16 -> M=lane-16, K = kk+{8..15, 24..31}.
__device__ __forceinline__ v16bf load_fragA(const __bf16* tile, int ld, int lane, int kk) {
  int r  = lane & 15;
  int k0 = kk + ((lane >> 4) << 3);
  const unsigned short* p = (const unsigned short*)(tile + (size_t)r * ld + k0);
  union { v16bf v; unsigned short s[16]; } f;
#pragma unroll
  for (int j = 0; j < 8; ++j) f.s[j] = p[j];
#pragma unroll
  for (int j = 0; j < 8; ++j) f.s[8 + j] = p[16 + j];
  return f.v;
}
// B fragment from LDS-staged tile, 32x16 bf16: lane<16 -> N=lane,
// K = kk+{0..7, 16..23}; lane>=16 -> N=lane-16, K = kk+{8..15, 24..31}.
__device__ __forceinline__ v16bf load_fragB_lds(const unsigned short* sB, int K, int lane, int kk) {
  int n  = lane & 15;
  int k0 = kk + ((lane >> 4) << 3);
  const unsigned short* p = sB + n * K + k0;
  union { v16bf v; unsigned short s[16]; } f;
#pragma unroll
  for (int j = 0; j < 8; ++j) f.s[j] = p[j];
#pragma unroll
  for (int j = 0; j < 8; ++j) f.s[8 + j] = p[16 + j];
  return f.v;
}

__global__ __launch_bounds__(128)
void wmma_gemm_bf16(const __bf16* __restrict__ A, int lda,
                    const __bf16* __restrict__ W, int ldb,
                    const float* __restrict__ bias,
                    float* __restrict__ C, int ldc, int K) {
  __shared__ alignas(16) unsigned short sB[16 * 256];   // B tile: 16 rows x K (<=256) bf16
  const int lane = threadIdx.x & 31;
  const int wave = threadIdx.x >> 5;
  const int tm = blockIdx.y * 4 + wave;   // 16-row tile index (per wave)
  const int tn = blockIdx.x;              // 16-col tile index (shared by block)
  const __bf16* At = A + (size_t)tm * 16 * lda;
  const __bf16* Wt = W + (size_t)tn * 16 * ldb;

  // Stage the 16 x K B tile into LDS once per block (shared by all 4 waves).
  // Uniform trip count across all threads -> no divergence, EXEC stays full.
  const int chunksPerRow = K >> 3;          // 16-byte chunks per row
  const int nChunks = chunksPerRow << 4;    // 16 rows
  for (int c = threadIdx.x; c < nChunks; c += 128) {
    int r  = c / chunksPerRow;
    int k8 = (c - r * chunksPerRow) << 3;
    const __bf16* src = Wt + (size_t)r * ldb + k8;
    unsigned short* dst = &sB[r * K + k8];
#if HAS_ASYNC_LDS
    async_copy_b128(src, dst);              // global_load_async_to_lds_b128
#else
    *(uint4*)dst = *(const uint4*)src;      // fallback: b128 load + ds_store_b128
#endif
  }
#if HAS_ASYNC_LDS
  wait_asynccnt0();                         // s_wait_asynccnt 0
#endif
  __syncthreads();

  v8f acc = {};
  for (int kk = 0; kk < K; kk += 32) {
    if (kk + 32 < K)
      __builtin_prefetch((const void*)(At + (size_t)(lane & 15) * lda + kk + 32), 0, 1);
    v16bf a = load_fragA(At, lda, lane, kk);
    v16bf b = load_fragB_lds(sB, K, lane, kk);
    acc = __builtin_amdgcn_wmma_f32_16x16x32_bf16(
        /*neg_a=*/false, a, /*neg_b=*/false, b,
        /*c_mod=*/(short)0, acc, /*reuse_a=*/false, /*reuse_b=*/false);
  }
  const int n  = tn * 16 + (lane & 15);
  const int m0 = tm * 16 + ((lane >> 4) << 3);
  const float bv = bias[n];
#pragma unroll
  for (int i = 0; i < 8; ++i)
    C[(size_t)(m0 + i) * ldc + n] = acc[i] + bv;
}

// ---------- prep kernels ----------

__global__ void prep_weights(const float* __restrict__ Wi, const float* __restrict__ Wh,
                             const float* __restrict__ bi, const float* __restrict__ bh,
                             const float* __restrict__ Wqg, const float* __restrict__ Wqp,
                             const float* __restrict__ Wrg, const float* __restrict__ Wrp,
                             __bf16* __restrict__ Wcat, float* __restrict__ bcat,
                             __bf16* __restrict__ wqg, __bf16* __restrict__ wqp,
                             __bf16* __restrict__ wrg, __bf16* __restrict__ wrp) {
  int tid = blockIdx.x * blockDim.x + threadIdx.x;
  const int NW = 512 * 256;
  if (tid < NW) {                     // Wcat[n][0:128]=Wi, [128:256]=Wh
    int n = tid >> 8, k = tid & 255;
    float v = (k < 128) ? Wi[n * 128 + k] : Wh[n * 128 + (k - 128)];
    store_bf(&Wcat[tid], v);
  } else if (tid < NW + 16384) {
    int i = tid - NW;               store_bf(&wqg[i], Wqg[i]);
  } else if (tid < NW + 2 * 16384) {
    int i = tid - (NW + 16384);     store_bf(&wqp[i], Wqp[i]);
  } else if (tid < NW + 3 * 16384) {
    int i = tid - (NW + 2 * 16384); store_bf(&wrg[i], Wrg[i]);
  } else if (tid < NW + 4 * 16384) {
    int i = tid - (NW + 3 * 16384); store_bf(&wrp[i], Wrp[i]);
  } else if (tid < NW + 4 * 16384 + 512) {
    int i = tid - (NW + 4 * 16384); bcat[i] = bi[i] + bh[i];
  }
}

// context (S,B,H) fp32 -> ctx_bf (B,S,H) bf16
__global__ void ctx_to_bf(const float* __restrict__ ctx, __bf16* __restrict__ out) {
  int tid = blockIdx.x * blockDim.x + threadIdx.x;   // B*S*H = 2^24
  int h = tid & 127;
  int s = (tid >> 7) & 255;
  int b = tid >> 15;
  store_bf(&out[tid], ctx[((size_t)s * BB + b) * HH + h]);
}

__global__ void init_state(const float* __restrict__ dec, const float* __restrict__ h0,
                           const float* __restrict__ c0, __bf16* __restrict__ Xcat,
                           float* __restrict__ hx, float* __restrict__ cx,
                           unsigned char* __restrict__ mask,
                           int* __restrict__ prev, int* __restrict__ fin) {
  int tid = blockIdx.x * blockDim.x + threadIdx.x;   // B*S = 131072
  if (tid < BB * SS) mask[tid] = 0;
  if (tid < BB * HH) {
    int b = tid >> 7, h = tid & 127;
    store_bf(&Xcat[(size_t)b * 256 + h],       dec[tid]);
    store_bf(&Xcat[(size_t)b * 256 + 128 + h], h0[tid]);
    hx[tid] = h0[tid];
    cx[tid] = c0[tid];
  }
  if (tid < BB) { prev[tid] = 0; fin[tid] = 1; }
}

// ---------- per-step kernels ----------

__global__ void lstm_cell(const float* __restrict__ gates,
                          float* __restrict__ cx, float* __restrict__ hx,
                          __bf16* __restrict__ Xcat) {
  int tid = blockIdx.x * blockDim.x + threadIdx.x;   // B*H
  int b = tid >> 7, h = tid & 127;
  const float* g = gates + (size_t)b * 512;
  float ig = sigmoidf(g[h]);
  float fg = sigmoidf(g[128 + h]);
  float cg = tanhf(g[256 + h]);
  float og = sigmoidf(g[384 + h]);
  float cy = fg * cx[tid] + ig * cg;
  float hy = og * tanhf(cy);
  cx[tid] = cy;
  hx[tid] = hy;
  store_bf(&Xcat[(size_t)b * 256 + 128 + h], hy);
}

__global__ __launch_bounds__(256)
void attn_g(const float* __restrict__ q, const float* __restrict__ eg,
            const float* __restrict__ vg, unsigned char* __restrict__ mask,
            const int* __restrict__ prev, int step,
            float* __restrict__ gl, __bf16* __restrict__ gl_bf) {
  __shared__ float sq[HH], sv[HH], su[SS], red[SS];
  const int b = blockIdx.x, tid = threadIdx.x;
  if (tid < HH) { sq[tid] = q[(size_t)b * HH + tid]; sv[tid] = vg[tid]; }
  if (tid == 0 && step > 0) {          // _apply_mask state update (has_prev)
    mask[(size_t)b * SS + prev[b]] = 1;
    mask[(size_t)b * SS] = 0;
  }
  __syncthreads();
  const int lane = tid & 31, wave = tid >> 5;
  const float* eb = eg + (size_t)b * SS * HH;
  for (int i = 0; i < 32; ++i) {
    int s = wave * 32 + i;
    const float* er = eb + (size_t)s * HH;
    float part = 0.f;
#pragma unroll
    for (int j = 0; j < 4; ++j) {
      int h = lane + 32 * j;
      part += sv[h] * tanhf(sq[h] + er[h]);
    }
#pragma unroll
    for (int off = 16; off; off >>= 1) part += __shfl_xor(part, off, 32);
    if (lane == 0) su[s] = mask[(size_t)b * SS + s] ? NEG_INF_F : part;
  }
  __syncthreads();
  // softmax over S
  red[tid] = su[tid]; __syncthreads();
  for (int off = 128; off; off >>= 1) {
    if (tid < off) red[tid] = fmaxf(red[tid], red[tid + off]);
    __syncthreads();
  }
  float mx = red[0]; __syncthreads();
  float ex = __expf(su[tid] - mx);
  red[tid] = ex; __syncthreads();
  for (int off = 128; off; off >>= 1) {
    if (tid < off) red[tid] += red[tid + off];
    __syncthreads();
  }
  float inv = 1.0f / red[0];
  su[tid] = ex * inv; __syncthreads();
  // g_l[b][h] = sum_s a[s]*e_g[b][s][h]
  if (tid < HH) {
    float acc = 0.f;
    for (int s = 0; s < SS; ++s) acc += su[s] * eb[(size_t)s * HH + tid];
    gl[(size_t)b * HH + tid] = acc;
    store_bf(&gl_bf[(size_t)b * HH + tid], acc);
  }
}

__global__ __launch_bounds__(256)
void attn_p(const float* __restrict__ q, const float* __restrict__ ep,
            const float* __restrict__ vp, const unsigned char* __restrict__ mask,
            const float* __restrict__ emb, int step,
            float* __restrict__ probs_out, float* __restrict__ sels_out,
            int* __restrict__ prev, int* __restrict__ fin,
            __bf16* __restrict__ Xcat) {
  __shared__ float sq[HH], sv[HH], slog[SS], red[SS], rv[SS];
  __shared__ int ri[SS];
  __shared__ int sel_sh;
  const int b = blockIdx.x, tid = threadIdx.x;
  if (tid < HH) { sq[tid] = q[(size_t)b * HH + tid]; sv[tid] = vp[tid]; }
  __syncthreads();
  const int lane = tid & 31, wave = tid >> 5;
  const float* eb = ep + (size_t)b * SS * HH;
  for (int i = 0; i < 32; ++i) {
    int s = wave * 32 + i;
    const float* er = eb + (size_t)s * HH;
    float part = 0.f;
#pragma unroll
    for (int j = 0; j < 4; ++j) {
      int h = lane + 32 * j;
      part += sv[h] * tanhf(sq[h] + er[h]);
    }
#pragma unroll
    for (int off = 16; off; off >>= 1) part += __shfl_xor(part, off, 32);
    if (lane == 0) {
      float lg = 10.0f * tanhf(part);  // C_EXPLORE * tanh(u)
      slog[s] = mask[(size_t)b * SS + s] ? NEG_INF_F : lg;
    }
  }
  __syncthreads();
  // softmax -> probs output
  red[tid] = slog[tid]; __syncthreads();
  for (int off = 128; off; off >>= 1) {
    if (tid < off) red[tid] = fmaxf(red[tid], red[tid + off]);
    __syncthreads();
  }
  float mx = red[0]; __syncthreads();
  float ex = __expf(slog[tid] - mx);
  red[tid] = ex; __syncthreads();
  for (int off = 128; off; off >>= 1) {
    if (tid < off) red[tid] += red[tid + off];
    __syncthreads();
  }
  float inv = 1.0f / red[0];
  probs_out[((size_t)step * BB + b) * SS + tid] = ex * inv;
  // deterministic Gumbel-max sampling
  unsigned h = hash3((unsigned)step, (unsigned)b, (unsigned)tid);
  float u01 = ((h >> 8) + 1u) * (1.0f / 16777217.0f);
  float gum = -logf(-logf(u01));
  rv[tid] = slog[tid] + gum; ri[tid] = tid; __syncthreads();
  for (int off = 128; off; off >>= 1) {
    if (tid < off && rv[tid + off] > rv[tid]) { rv[tid] = rv[tid + off]; ri[tid] = ri[tid + off]; }
    __syncthreads();
  }
  if (tid == 0) {
    int f = fin[b];
    int idx = ri[0] * f;
    fin[b] = f * (idx != 0 ? 1 : 0);
    prev[b] = idx;
    sels_out[(size_t)step * BB + b] = (float)idx;
    sel_sh = idx;
  }
  __syncthreads();
  if (tid < EE)   // dec_in_new = embedded_inputs[idx, b, :]
    store_bf(&Xcat[(size_t)b * 256 + tid], emb[((size_t)sel_sh * BB + b) * EE + tid]);
}

__global__ void finalize(const float* __restrict__ hx, const float* __restrict__ cx,
                         float* __restrict__ oh, float* __restrict__ oc) {
  int tid = blockIdx.x * blockDim.x + threadIdx.x;
  if (tid < BB * HH) { oh[tid] = hx[tid]; oc[tid] = cx[tid]; }
}

// ---------- host ----------

extern "C" void kernel_launch(void* const* d_in, const int* in_sizes, int n_in,
                              void* d_out, int out_size, void* d_ws, size_t ws_size,
                              hipStream_t stream) {
  (void)in_sizes; (void)n_in; (void)out_size; (void)ws_size;
  const float* dec = (const float*)d_in[0];
  const float* emb = (const float*)d_in[1];
  const float* h0  = (const float*)d_in[2];
  const float* c0  = (const float*)d_in[3];
  const float* ctx = (const float*)d_in[4];
  const float* Wi  = (const float*)d_in[5];
  const float* bi  = (const float*)d_in[6];
  const float* Wh  = (const float*)d_in[7];
  const float* bh  = (const float*)d_in[8];
  const float* Wqg = (const float*)d_in[9];
  const float* bqg = (const float*)d_in[10];
  const float* Wrg = (const float*)d_in[11];
  const float* brg = (const float*)d_in[12];
  const float* vg  = (const float*)d_in[13];
  const float* Wqp = (const float*)d_in[14];
  const float* bqp = (const float*)d_in[15];
  const float* Wrp = (const float*)d_in[16];
  const float* brp = (const float*)d_in[17];
  const float* vp  = (const float*)d_in[18];

  char* ws = (char*)d_ws;
  size_t off = 0;
  auto nxt = [&](size_t bytes) { size_t o = off; off += (bytes + 255) & ~(size_t)255; return o; };
  float*  eg    = (float*)(ws + nxt((size_t)BB * SS * HH * 4));   // (B,S,H)
  float*  ep    = (float*)(ws + nxt((size_t)BB * SS * HH * 4));
  __bf16* ctxbf = (__bf16*)(ws + nxt((size_t)BB * SS * HH * 2));
  __bf16* Wcat  = (__bf16*)(ws + nxt((size_t)512 * 256 * 2));
  float*  bcat  = (float*)(ws + nxt(512 * 4));
  __bf16* wqg   = (__bf16*)(ws + nxt(16384 * 2));
  __bf16* wqp   = (__bf16*)(ws + nxt(16384 * 2));
  __bf16* wrg   = (__bf16*)(ws + nxt(16384 * 2));
  __bf16* wrp   = (__bf16*)(ws + nxt(16384 * 2));
  __bf16* Xcat  = (__bf16*)(ws + nxt((size_t)512 * 256 * 2));     // [dec_in | h] bf16
  float*  gates = (float*)(ws + nxt((size_t)512 * 512 * 4));
  float*  hx    = (float*)(ws + nxt(65536 * 4));
  float*  cx    = (float*)(ws + nxt(65536 * 4));
  float*  qbuf  = (float*)(ws + nxt(65536 * 4));
  float*  gl    = (float*)(ws + nxt(65536 * 4));
  __bf16* glbf  = (__bf16*)(ws + nxt(65536 * 2));
  unsigned char* mask = (unsigned char*)(ws + nxt(131072));
  int*    prev  = (int*)(ws + nxt(512 * 4));
  int*    fin   = (int*)(ws + nxt(512 * 4));

  float* out = (float*)d_out;
  float* probs_out = out;                               // T*B*S
  float* sels_out  = out + (size_t)T_STEPS * BB * SS;   // T*B
  float* hx_out    = sels_out + (size_t)T_STEPS * BB;   // B*H
  float* cx_out    = hx_out + (size_t)BB * HH;          // B*H

  // one-time prep (re-run every call for determinism)
  prep_weights<<<770, 256, 0, stream>>>(Wi, Wh, bi, bh, Wqg, Wqp, Wrg, Wrp,
                                        Wcat, bcat, wqg, wqp, wrg, wrp);
  ctx_to_bf<<<65536, 256, 0, stream>>>(ctx, ctxbf);
  // e_g / e_p: (B*S,128) x (128,128)^T
  wmma_gemm_bf16<<<dim3(HH / 16, (BB * SS) / 64), 128, 0, stream>>>(
      ctxbf, HH, wrg, HH, brg, eg, HH, HH);
  wmma_gemm_bf16<<<dim3(HH / 16, (BB * SS) / 64), 128, 0, stream>>>(
      ctxbf, HH, wrp, HH, brp, ep, HH, HH);
  init_state<<<512, 256, 0, stream>>>(dec, h0, c0, Xcat, hx, cx, mask, prev, fin);

  for (int t = 0; t < T_STEPS; ++t) {
    // gates = [dec_in|h] @ [Wi|Wh]^T + (bi+bh)   (512 x 512 x 256)
    wmma_gemm_bf16<<<dim3(512 / 16, 512 / 64), 128, 0, stream>>>(
        Xcat, 256, Wcat, 256, bcat, gates, 512, 256);
    lstm_cell<<<256, 256, 0, stream>>>(gates, cx, hx, Xcat);
    // q_g = h @ Wq_g^T + bq_g   (A = hy half of Xcat)
    wmma_gemm_bf16<<<dim3(8, 8), 128, 0, stream>>>(
        Xcat + 128, 256, wqg, 128, bqg, qbuf, 128, 128);
    attn_g<<<512, 256, 0, stream>>>(qbuf, eg, vg, mask, prev, t, gl, glbf);
    // q_p = g_l @ Wq_p^T + bq_p
    wmma_gemm_bf16<<<dim3(8, 8), 128, 0, stream>>>(
        glbf, 128, wqp, 128, bqp, qbuf, 128, 128);
    attn_p<<<512, 256, 0, stream>>>(qbuf, ep, vp, mask, emb, t,
                                    probs_out, sels_out, prev, fin, Xcat);
  }
  finalize<<<256, 256, 0, stream>>>(hx, cx, hx_out, cx_out);
}
